// TemporalGraphAttention_28080496181759
// MI455X (gfx1250) — matrix operand order repaired and checked by reading, not compiled, not run on hardware
//
#include <hip/hip_runtime.h>
#include <hip/hip_bf16.h>

// ---------------------------------------------------------------------------
// Types for CDNA5 WMMA
// ---------------------------------------------------------------------------
typedef __attribute__((ext_vector_type(16))) __bf16 v16bf;
typedef __attribute__((ext_vector_type(8)))  float  v8f;

#define HEADS1 8
#define HEADS3 4
#define HIDC   128
#define WID1   1024   // HID*H1
#define WID3   512    // HID*H3
#define KIN    64     // 40 padded to 64
#define NGRP   32
#define NEG_SLOPE 0.2f

// ---- CDNA5 async global->LDS path (guarded; sync fallback is known-good) ----
#if defined(__HIP_DEVICE_COMPILE__) && __has_builtin(__builtin_amdgcn_global_load_async_to_lds_b128)
#define ASYNC_LDS 1
typedef int v4i_async __attribute__((vector_size(16)));
typedef __attribute__((address_space(1))) v4i_async* gasync_p;
typedef __attribute__((address_space(3))) v4i_async* lasync_p;
#else
#define ASYNC_LDS 0
#endif

__device__ inline void wait_async0() {
#if ASYNC_LDS
#if __has_builtin(__builtin_amdgcn_s_wait_asynccnt)
    __builtin_amdgcn_s_wait_asynccnt(0);
#else
    asm volatile("s_wait_asynccnt 0x0" ::: "memory");
#endif
#endif
}

// Copy 16 bytes global -> LDS (async when available), or zero-fill when !valid
__device__ inline void stage_b128(const unsigned short* __restrict__ g,
                                  unsigned short* l, bool valid) {
    if (valid) {
#if ASYNC_LDS
        __builtin_amdgcn_global_load_async_to_lds_b128((gasync_p)g, (lasync_p)l, 0, 0);
#else
        *(uint4*)l = *(const uint4*)g;
#endif
    } else {
        *(uint4*)l = make_uint4(0u, 0u, 0u, 0u);
    }
}

// ---------------------------------------------------------------------------
// Helpers
// ---------------------------------------------------------------------------
__device__ inline unsigned short f32_to_bf16_raw(float f) {
    unsigned int u = __float_as_uint(f);
    unsigned int r = u + 0x7FFFu + ((u >> 16) & 1u);   // round-to-nearest-even
    return (unsigned short)(r >> 16);
}

__device__ inline void atomicAddF(float* p, float v) {
    __hip_atomic_fetch_add(p, v, __ATOMIC_RELAXED, __HIP_MEMORY_SCOPE_AGENT);
}

__device__ inline void atomicMaxF(float* addr, float val) {
    unsigned int* ua = (unsigned int*)addr;
    unsigned int old = *ua;
    while (__uint_as_float(old) < val) {
        unsigned int assumed = old;
        old = atomicCAS(ua, assumed, __float_as_uint(val));
        if (old == assumed) break;
    }
}

__device__ inline void edge_sd(const int* __restrict__ ei, int E, int i, int& s, int& d) {
    if (i < E) { s = ei[i]; d = ei[E + i]; }
    else       { s = i - E; d = i - E; }        // self loops appended
}

// ---------------------------------------------------------------------------
// Generic fills
// ---------------------------------------------------------------------------
__global__ void fill_f32(float* p, float v, int n) {
    int i = blockIdx.x * blockDim.x + threadIdx.x;
    if (i < n) p[i] = v;
}
__global__ void fill_i32(int* p, int v, int n) {
    int i = blockIdx.x * blockDim.x + threadIdx.x;
    if (i < n) p[i] = v;
}

// ---------------------------------------------------------------------------
// Weight f32 (K x M) -> bf16 TRANSPOSED (M x Kpad, zero-padded K)
// Transposed layout makes GEMM B-tile staging contiguous 16B chunks.
// ---------------------------------------------------------------------------
__global__ void pack_weight_bf16_T(const float* __restrict__ w, unsigned short* __restrict__ out,
                                   int K, int Kpad, int M) {
    size_t id = (size_t)blockIdx.x * blockDim.x + threadIdx.x;
    if (id >= (size_t)M * Kpad) return;
    int m = (int)(id / Kpad), k = (int)(id % Kpad);
    out[id] = (k < K) ? f32_to_bf16_raw(w[(size_t)k * M + m]) : (unsigned short)0;
}

// ---------------------------------------------------------------------------
// Time embedding + xin build (bf16, N x 64, cols 40..63 zero)
// ---------------------------------------------------------------------------
__global__ void time_embed_xin(const float* __restrict__ x,
                               const float* __restrict__ tW1, const float* __restrict__ tb1,
                               const float* __restrict__ tW2, const float* __restrict__ tb2,
                               unsigned short* __restrict__ xinb, int n) {
    int i = blockIdx.x * blockDim.x + threadIdx.x;
    if (i >= n) return;
    float hour = x[i * 8 + 2];
    float t1[32];
    #pragma unroll
    for (int j = 0; j < 32; ++j) {
        float v = hour * tW1[j] + tb1[j];
        t1[j] = v > 0.f ? v : 0.f;
    }
    #pragma unroll
    for (int k = 0; k < 8; ++k) xinb[i * 64 + k] = f32_to_bf16_raw(x[i * 8 + k]);
    for (int j = 0; j < 32; ++j) {
        float acc = tb2[j];
        #pragma unroll
        for (int k = 0; k < 32; ++k) acc += t1[k] * tW2[k * 32 + j];
        xinb[i * 64 + 8 + j] = f32_to_bf16_raw(acc);
    }
    #pragma unroll
    for (int k = 40; k < 64; ++k) xinb[i * 64 + k] = 0;
}

// ---------------------------------------------------------------------------
// bf16 WMMA GEMM:  C(f32, rows x M) = A(bf16, rows x K) @ B, with B given
// TRANSPOSED: BT(bf16, M x K).  Block: 256 threads (8 waves).
// Tile 128(rows) x 128(cols), K-step 32. Wave w owns rows [w*16, w*16+16),
// all 8 column sub-tiles -> 8 WMMA per wave per K-step.
// ---------------------------------------------------------------------------
#define TILE_M 128
#define TILE_N 128
#define TILE_K 32
#define LDS_STRIDE 40   // bf16 elems per A-row / B-column (32 + 8 pad)

__global__ __launch_bounds__(256)
void gemm_bf16_wmma(const unsigned short* __restrict__ A,
                    const unsigned short* __restrict__ BT,
                    float* __restrict__ C,
                    int rows, int K, int M) {
    __shared__ __align__(16) unsigned short As[TILE_M * LDS_STRIDE];
    __shared__ __align__(16) unsigned short Bs[TILE_N * LDS_STRIDE];

    const int tid  = threadIdx.x;
    const int wave = tid >> 5;
    const int lane = tid & 31;
    const int row0 = blockIdx.x * TILE_M;
    const int col0 = blockIdx.y * TILE_N;

    v8f acc[8];
    #pragma unroll
    for (int t = 0; t < 8; ++t)
        #pragma unroll
        for (int q = 0; q < 8; ++q) acc[t][q] = 0.f;

    const int m_a  = lane & 15;
    const int hsel = lane >> 4;

    for (int kb = 0; kb < K; kb += TILE_K) {
        // ---- stage A tile (128 rows x 32 bf16 = 512 chunks of 16B)
        #pragma unroll
        for (int cc = 0; cc < 2; ++cc) {
            int chunk = tid + cc * 256;       // 0..511
            int r = chunk >> 2;               // row 0..127
            int h = chunk & 3;                // 16B chunk within 64B row
            int grow = row0 + r;
            stage_b128(A + (size_t)grow * K + kb + h * 8,
                       &As[r * LDS_STRIDE + h * 8], grow < rows);
        }
        // ---- stage B tile (128 cols x 32 bf16, from BT rows = B columns)
        #pragma unroll
        for (int cc = 0; cc < 2; ++cc) {
            int chunk = tid + cc * 256;
            int c = chunk >> 2;               // column 0..127
            int h = chunk & 3;
            stage_b128(BT + (size_t)(col0 + c) * K + kb + h * 8,
                       &Bs[c * LDS_STRIDE + h * 8], true);
        }
        wait_async0();
        __syncthreads();

        // ---- A fragment (16x32 bf16 ISA layout); reused across 8 WMMAs
        union { v16bf v; unsigned int u[8]; } afrag;
        {
            int rbase = (wave * 16 + m_a) * LDS_STRIDE;
            #pragma unroll
            for (int vv = 0; vv < 8; ++vv) {
                int k = ((vv < 4) ? (2 * vv) : (8 + 2 * vv)) + 8 * hsel;
                afrag.u[vv] = *(const unsigned int*)(&As[rbase + k]);
            }
        }
        // ---- 8 column tiles
        #pragma unroll
        for (int t = 0; t < 8; ++t) {
            union { v16bf v; unsigned int u[8]; } bfrag;
            int cb = t * 16 + (lane & 15);
            #pragma unroll
            for (int vv = 0; vv < 8; ++vv) {
                int k = ((vv < 4) ? (2 * vv) : (8 + 2 * vv)) + 8 * hsel;
                bfrag.u[vv] = *(const unsigned int*)(&Bs[cb * LDS_STRIDE + k]);
            }
            acc[t] = __builtin_amdgcn_wmma_f32_16x16x32_bf16(
                false, afrag.v, false, bfrag.v, (short)0, acc[t], false, false);
        }
        __syncthreads();
    }

    // ---- epilogue: C layout (lane = N, vgpr + 8*hsel = M)
    int n = lane & 15;
    #pragma unroll
    for (int t = 0; t < 8; ++t) {
        int col = col0 + t * 16 + n;
        #pragma unroll
        for (int vv = 0; vv < 8; ++vv) {
            int row = row0 + wave * 16 + vv + 8 * hsel;
            if (row < rows) C[(size_t)row * M + col] = acc[t][vv];
        }
    }
}

// ---------------------------------------------------------------------------
// Attention logits: al_s[n,h] = <h[n,h,:], a_src[h,:]> (HID=128), one wave/head
// ---------------------------------------------------------------------------
__global__ void attn_logits(const float* __restrict__ h,
                            const float* __restrict__ a_src, const float* __restrict__ a_dst,
                            float* __restrict__ als, float* __restrict__ ald, int heads) {
    int node = blockIdx.x;
    int hh   = threadIdx.x >> 5;
    int lane = threadIdx.x & 31;
    if (hh >= heads) return;
    const float* hp = h + (size_t)node * heads * HIDC + hh * HIDC;
    const float* as = a_src + hh * HIDC;
    const float* ad = a_dst + hh * HIDC;
    float ss = 0.f, sd = 0.f;
    #pragma unroll
    for (int j = 0; j < 4; ++j) {
        float hv = hp[lane + 32 * j];
        ss += hv * as[lane + 32 * j];
        sd += hv * ad[lane + 32 * j];
    }
    #pragma unroll
    for (int off = 16; off; off >>= 1) {
        ss += __shfl_down(ss, off, 32);
        sd += __shfl_down(sd, off, 32);
    }
    if (lane == 0) { als[node * heads + hh] = ss; ald[node * heads + hh] = sd; }
}

// ---------------------------------------------------------------------------
// Edge pass 1: e = leakyrelu(als[src]+ald[dst]); segment max into m[dst]
// ---------------------------------------------------------------------------
__global__ void edge_max_kernel(const int* __restrict__ ei, int E, int ET, int heads,
                                const float* __restrict__ als, const float* __restrict__ ald,
                                float* __restrict__ ebuf, float* __restrict__ m) {
    int gid = blockIdx.x * blockDim.x + threadIdx.x;
    if (gid >= ET * heads) return;
    int i = gid / heads, hh = gid - i * heads;
    int s, d; edge_sd(ei, E, i, s, d);
    float e = als[s * heads + hh] + ald[d * heads + hh];
    e = e > 0.f ? e : NEG_SLOPE * e;
    ebuf[gid] = e;
    atomicMaxF(&m[d * heads + hh], e);
}

// ---------------------------------------------------------------------------
// Edge pass 2: p = exp(e - m[dst]); segment sum into den[dst]
// ---------------------------------------------------------------------------
__global__ void edge_p_kernel(const int* __restrict__ ei, int E, int ET, int heads,
                              const float* __restrict__ m,
                              float* __restrict__ ebuf, float* __restrict__ den) {
    int gid = blockIdx.x * blockDim.x + threadIdx.x;
    if (gid >= ET * heads) return;
    int i = gid / heads, hh = gid - i * heads;
    int s, d; edge_sd(ei, E, i, s, d);
    float p = __expf(ebuf[gid] - m[d * heads + hh]);
    ebuf[gid] = p;
    atomicAddF(&den[d * heads + hh], p);
}

// ---------------------------------------------------------------------------
// Edge pass 3: agg[dst] += (p/den[dst]) * h[src]   (one wave per head)
// ---------------------------------------------------------------------------
__global__ void edge_agg_kernel(const int* __restrict__ ei, int E, int heads,
                                const float* __restrict__ h, const float* __restrict__ p,
                                const float* __restrict__ den, float* __restrict__ agg) {
    int i    = blockIdx.x;
    int hh   = threadIdx.x >> 5;
    int lane = threadIdx.x & 31;
    int s, d; edge_sd(ei, E, i, s, d);
    float alpha = p[i * heads + hh] / (den[d * heads + hh] + 1e-16f);
    const float* hs = h   + (size_t)s * heads * HIDC + hh * HIDC + lane * 4;
    float*       od = agg + (size_t)d * heads * HIDC + hh * HIDC + lane * 4;
    float4 hv = *(const float4*)hs;
    atomicAddF(od + 0, alpha * hv.x);
    atomicAddF(od + 1, alpha * hv.y);
    atomicAddF(od + 2, alpha * hv.z);
    atomicAddF(od + 3, alpha * hv.w);
}

// ---------------------------------------------------------------------------
// Column statistics for BatchNorm (one block per column)
// ---------------------------------------------------------------------------
__global__ void colstats(const float* __restrict__ x, int rows, int M,
                         float* __restrict__ mean, float* __restrict__ rstd) {
    __shared__ float s1[256], s2[256];
    int j = blockIdx.x;
    float a = 0.f, b = 0.f;
    for (int r = threadIdx.x; r < rows; r += 256) {
        float v = x[(size_t)r * M + j];
        a += v; b += v * v;
    }
    s1[threadIdx.x] = a; s2[threadIdx.x] = b;
    __syncthreads();
    for (int o = 128; o; o >>= 1) {
        if (threadIdx.x < o) {
            s1[threadIdx.x] += s1[threadIdx.x + o];
            s2[threadIdx.x] += s2[threadIdx.x + o];
        }
        __syncthreads();
    }
    if (threadIdx.x == 0) {
        float mu  = s1[0] / rows;
        float var = s2[0] / rows - mu * mu;
        mean[j] = mu;
        rstd[j] = rsqrtf(var + 1e-5f);
    }
}

// ---------------------------------------------------------------------------
// x = relu(bn(agg)) [+ skip + skip_b]; write f32 and optional bf16
// ---------------------------------------------------------------------------
__global__ void bn_relu_skip(const float* __restrict__ agg,
                             const float* __restrict__ mean, const float* __restrict__ rstd,
                             const float* __restrict__ g, const float* __restrict__ b,
                             const float* __restrict__ skip, const float* __restrict__ skipb,
                             float* __restrict__ xout, unsigned short* __restrict__ xbout,
                             int rows, int M) {
    size_t id = (size_t)blockIdx.x * blockDim.x + threadIdx.x;
    if (id >= (size_t)rows * M) return;
    int j = (int)(id % M);
    float v = (agg[id] - mean[j]) * rstd[j] * g[j] + b[j];
    v = v > 0.f ? v : 0.f;
    if (skip)  v += skip[id];
    if (skipb) v += skipb[j];
    xout[id] = v;
    if (xbout) xbout[id] = f32_to_bf16_raw(v);
}

// ---------------------------------------------------------------------------
// Pooling
// ---------------------------------------------------------------------------
__global__ void pool_count(const int* __restrict__ batch, int* __restrict__ cnt, int n) {
    int i = blockIdx.x * blockDim.x + threadIdx.x;
    if (i < n) atomicAdd(&cnt[batch[i]], 1);
}
__global__ void pool_acc(const float* __restrict__ x3, const int* __restrict__ batch,
                         float* __restrict__ psum, float* __restrict__ pmax, int n, int M) {
    size_t id = (size_t)blockIdx.x * blockDim.x + threadIdx.x;
    if (id >= (size_t)n * M) return;
    int node = (int)(id / M), c = (int)(id % M);
    int g = batch[node];
    float v = x3[id];
    atomicAddF(&psum[g * M + c], v);
    atomicMaxF(&pmax[g * M + c], v);
}
__global__ void pool_build(const float* __restrict__ psum, const float* __restrict__ pmax,
                           const int* __restrict__ cnt, float* __restrict__ pooled, int M) {
    int id = blockIdx.x * blockDim.x + threadIdx.x;   // NGRP * 2M
    if (id >= NGRP * 2 * M) return;
    int g = id / (2 * M), j = id % (2 * M);
    if (j < M) {
        float c = (float)cnt[g];
        pooled[id] = psum[g * M + j] / (c > 1.f ? c : 1.f);
    } else {
        pooled[id] = pmax[g * M + (j - M)];
    }
}

// ---------------------------------------------------------------------------
// Tiny MLP head kernels (32 rows)
// ---------------------------------------------------------------------------
__global__ void mlp_gemm(const float* __restrict__ A, const float* __restrict__ W,
                         const float* __restrict__ bias, float* __restrict__ C,
                         int rows, int K, int M, int relu) {
    int id = blockIdx.x * blockDim.x + threadIdx.x;
    if (id >= rows * M) return;
    int r = id / M, c = id - r * M;
    float acc = bias[c];
    for (int k = 0; k < K; ++k) acc += A[r * K + k] * W[k * M + c];
    if (relu && acc < 0.f) acc = 0.f;
    C[id] = acc;
}
__global__ void mlp_bn(float* __restrict__ x, const float* __restrict__ g,
                       const float* __restrict__ b, int rows, int M) {
    int j = blockIdx.x * blockDim.x + threadIdx.x;
    if (j >= M) return;
    float s = 0.f, s2 = 0.f;
    for (int r = 0; r < rows; ++r) { float v = x[r * M + j]; s += v; s2 += v * v; }
    float mu = s / rows, var = s2 / rows - mu * mu;
    float rs = rsqrtf(var + 1e-5f);
    for (int r = 0; r < rows; ++r) x[r * M + j] = g[j] * (x[r * M + j] - mu) * rs + b[j];
}

// ---------------------------------------------------------------------------
// Host launcher
// ---------------------------------------------------------------------------
extern "C" void kernel_launch(void* const* d_in, const int* in_sizes, int n_in,
                              void* d_out, int out_size, void* d_ws, size_t ws_size,
                              hipStream_t stream) {
    const float* x    = (const float*)d_in[0];
    const int*   ei   = (const int*)  d_in[1];
    const int*   batch= (const int*)  d_in[2];
    const float* tW1  = (const float*)d_in[3];
    const float* tb1  = (const float*)d_in[4];
    const float* tW2  = (const float*)d_in[5];
    const float* tb2  = (const float*)d_in[6];
    const float* g1_W = (const float*)d_in[7];
    const float* g1_as= (const float*)d_in[8];
    const float* g1_ad= (const float*)d_in[9];
    /* g1_b/g2_b/g3_b (d_in[10/14/18]): per-column shift cancels in BN (exact) */
    const float* g2_W = (const float*)d_in[11];
    const float* g2_as= (const float*)d_in[12];
    const float* g2_ad= (const float*)d_in[13];
    const float* g3_W = (const float*)d_in[15];
    const float* g3_as= (const float*)d_in[16];
    const float* g3_ad= (const float*)d_in[17];
    const float* sk1_W= (const float*)d_in[19];
    const float* sk1_b= (const float*)d_in[20];
    const float* sk2_W= (const float*)d_in[21];
    const float* sk2_b= (const float*)d_in[22];
    const float* bn1_g= (const float*)d_in[23];
    const float* bn1_b= (const float*)d_in[24];
    const float* bn2_g= (const float*)d_in[25];
    const float* bn2_b= (const float*)d_in[26];
    const float* bn3_g= (const float*)d_in[27];
    const float* bn3_b= (const float*)d_in[28];
    const float* pW1  = (const float*)d_in[29];
    const float* pb1  = (const float*)d_in[30];
    const float* pbn1g= (const float*)d_in[31];
    const float* pbn1b= (const float*)d_in[32];
    const float* pW2  = (const float*)d_in[33];
    const float* pb2  = (const float*)d_in[34];
    const float* pbn2g= (const float*)d_in[35];
    const float* pbn2b= (const float*)d_in[36];
    const float* pW3  = (const float*)d_in[37];
    const float* pb3  = (const float*)d_in[38];
    const float* pW4  = (const float*)d_in[39];
    const float* pb4  = (const float*)d_in[40];

    const int N  = in_sizes[2];           // 10000
    const int E  = in_sizes[1] / 2;       // 80000
    const int ET = E + N;                 // with self loops

    // ---- workspace carve-out (256B aligned) ----
    char* ws = (char*)d_ws;
    size_t off = 0;
    auto carve = [&](size_t bytes) -> char* {
        char* p = ws + off;
        off = (off + bytes + 255) & ~(size_t)255;
        return p;
    };
    unsigned short* xinb   = (unsigned short*)carve((size_t)N * KIN * 2);
    unsigned short* wg1    = (unsigned short*)carve((size_t)KIN * WID1 * 2);   // BT: 1024 x 64
    unsigned short* wsk1   = (unsigned short*)carve((size_t)KIN * WID1 * 2);
    unsigned short* wg2    = (unsigned short*)carve((size_t)WID1 * WID1 * 2);  // BT: 1024 x 1024
    unsigned short* wsk2   = (unsigned short*)carve((size_t)WID1 * WID1 * 2);
    unsigned short* wg3    = (unsigned short*)carve((size_t)WID1 * WID3 * 2);  // BT: 512 x 1024
    float*          bh     = (float*)         carve((size_t)N * WID1 * 4);
    float*          bs     = (float*)         carve((size_t)N * WID1 * 4);
    float*          bagg   = (float*)         carve((size_t)N * WID1 * 4);
    float*          bx     = (float*)         carve((size_t)N * WID1 * 4);
    unsigned short* bxb    = (unsigned short*)carve((size_t)N * WID1 * 2);
    float*          als    = (float*)         carve((size_t)N * HEADS1 * 4);
    float*          ald    = (float*)         carve((size_t)N * HEADS1 * 4);
    float*          mbuf   = (float*)         carve((size_t)N * HEADS1 * 4);
    float*          den    = (float*)         carve((size_t)N * HEADS1 * 4);
    float*          ebuf   = (float*)         carve((size_t)ET * HEADS1 * 4);
    float*          cmean  = (float*)         carve(WID1 * 4);
    float*          crstd  = (float*)         carve(WID1 * 4);
    float*          psum   = (float*)         carve((size_t)NGRP * WID3 * 4);
    float*          pmax   = (float*)         carve((size_t)NGRP * WID3 * 4);
    int*            cnt    = (int*)           carve(NGRP * 4);
    float*          pooled = (float*)         carve((size_t)NGRP * 2 * WID3 * 4);
    float*          mh1    = (float*)         carve((size_t)NGRP * 256 * 4);
    float*          mh2    = (float*)         carve((size_t)NGRP * 128 * 4);
    float*          mh3    = (float*)         carve((size_t)NGRP * 64 * 4);
    (void)ws_size; (void)n_in; (void)out_size;

    const int TPB = 256;
    auto blocks = [](size_t n, int tpb) { return (int)((n + tpb - 1) / tpb); };
    const int gemmMB = (N + TILE_M - 1) / TILE_M;

    // ---- 0) weights -> bf16, transposed (M x Kpad) ----
    pack_weight_bf16_T<<<blocks((size_t)WID1 * KIN, TPB), TPB, 0, stream>>>(g1_W,  wg1,  40,   KIN,  WID1);
    pack_weight_bf16_T<<<blocks((size_t)WID1 * KIN, TPB), TPB, 0, stream>>>(sk1_W, wsk1, 40,   KIN,  WID1);
    pack_weight_bf16_T<<<blocks((size_t)WID1 * WID1, TPB), TPB, 0, stream>>>(g2_W,  wg2,  WID1, WID1, WID1);
    pack_weight_bf16_T<<<blocks((size_t)WID1 * WID1, TPB), TPB, 0, stream>>>(sk2_W, wsk2, WID1, WID1, WID1);
    pack_weight_bf16_T<<<blocks((size_t)WID3 * WID1, TPB), TPB, 0, stream>>>(g3_W,  wg3,  WID1, WID1, WID3);

    // ---- 1) time embedding + xin (bf16, padded) ----
    time_embed_xin<<<blocks(N, TPB), TPB, 0, stream>>>(x, tW1, tb1, tW2, tb2, xinb, N);

    // ============== Layer 1 ==============
    gemm_bf16_wmma<<<dim3(gemmMB, WID1 / TILE_N), TPB, 0, stream>>>(xinb, wg1,  bh, N, KIN, WID1);
    gemm_bf16_wmma<<<dim3(gemmMB, WID1 / TILE_N), TPB, 0, stream>>>(xinb, wsk1, bs, N, KIN, WID1);
    attn_logits<<<N, HEADS1 * 32, 0, stream>>>(bh, g1_as, g1_ad, als, ald, HEADS1);
    fill_f32<<<blocks((size_t)N * HEADS1, TPB), TPB, 0, stream>>>(mbuf, -1e38f, N * HEADS1);
    fill_f32<<<blocks((size_t)N * HEADS1, TPB), TPB, 0, stream>>>(den, 0.f, N * HEADS1);
    fill_f32<<<blocks((size_t)N * WID1, TPB), TPB, 0, stream>>>(bagg, 0.f, N * WID1);
    edge_max_kernel<<<blocks((size_t)ET * HEADS1, TPB), TPB, 0, stream>>>(ei, E, ET, HEADS1, als, ald, ebuf, mbuf);
    edge_p_kernel<<<blocks((size_t)ET * HEADS1, TPB), TPB, 0, stream>>>(ei, E, ET, HEADS1, mbuf, ebuf, den);
    edge_agg_kernel<<<ET, HEADS1 * 32, 0, stream>>>(ei, E, HEADS1, bh, ebuf, den, bagg);
    colstats<<<WID1, 256, 0, stream>>>(bagg, N, WID1, cmean, crstd);
    bn_relu_skip<<<blocks((size_t)N * WID1, TPB), TPB, 0, stream>>>(
        bagg, cmean, crstd, bn1_g, bn1_b, bs, sk1_b, bx, bxb, N, WID1);

    // ============== Layer 2 ==============
    gemm_bf16_wmma<<<dim3(gemmMB, WID1 / TILE_N), TPB, 0, stream>>>(bxb, wg2,  bh, N, WID1, WID1);
    gemm_bf16_wmma<<<dim3(gemmMB, WID1 / TILE_N), TPB, 0, stream>>>(bxb, wsk2, bs, N, WID1, WID1);
    attn_logits<<<N, HEADS1 * 32, 0, stream>>>(bh, g2_as, g2_ad, als, ald, HEADS1);
    fill_f32<<<blocks((size_t)N * HEADS1, TPB), TPB, 0, stream>>>(mbuf, -1e38f, N * HEADS1);
    fill_f32<<<blocks((size_t)N * HEADS1, TPB), TPB, 0, stream>>>(den, 0.f, N * HEADS1);
    fill_f32<<<blocks((size_t)N * WID1, TPB), TPB, 0, stream>>>(bagg, 0.f, N * WID1);
    edge_max_kernel<<<blocks((size_t)ET * HEADS1, TPB), TPB, 0, stream>>>(ei, E, ET, HEADS1, als, ald, ebuf, mbuf);
    edge_p_kernel<<<blocks((size_t)ET * HEADS1, TPB), TPB, 0, stream>>>(ei, E, ET, HEADS1, mbuf, ebuf, den);
    edge_agg_kernel<<<ET, HEADS1 * 32, 0, stream>>>(ei, E, HEADS1, bh, ebuf, den, bagg);
    colstats<<<WID1, 256, 0, stream>>>(bagg, N, WID1, cmean, crstd);
    bn_relu_skip<<<blocks((size_t)N * WID1, TPB), TPB, 0, stream>>>(
        bagg, cmean, crstd, bn2_g, bn2_b, bs, sk2_b, bx, bxb, N, WID1);

    // ============== Layer 3 (heads=4, width 512, no skip) ==============
    gemm_bf16_wmma<<<dim3(gemmMB, WID3 / TILE_N), TPB, 0, stream>>>(bxb, wg3, bh, N, WID1, WID3);
    attn_logits<<<N, HEADS3 * 32, 0, stream>>>(bh, g3_as, g3_ad, als, ald, HEADS3);
    fill_f32<<<blocks((size_t)N * HEADS3, TPB), TPB, 0, stream>>>(mbuf, -1e38f, N * HEADS3);
    fill_f32<<<blocks((size_t)N * HEADS3, TPB), TPB, 0, stream>>>(den, 0.f, N * HEADS3);
    fill_f32<<<blocks((size_t)N * WID3, TPB), TPB, 0, stream>>>(bagg, 0.f, N * WID3);
    edge_max_kernel<<<blocks((size_t)ET * HEADS3, TPB), TPB, 0, stream>>>(ei, E, ET, HEADS3, als, ald, ebuf, mbuf);
    edge_p_kernel<<<blocks((size_t)ET * HEADS3, TPB), TPB, 0, stream>>>(ei, E, ET, HEADS3, mbuf, ebuf, den);
    edge_agg_kernel<<<ET, HEADS3 * 32, 0, stream>>>(ei, E, HEADS3, bh, ebuf, den, bagg);
    colstats<<<WID3, 256, 0, stream>>>(bagg, N, WID3, cmean, crstd);
    bn_relu_skip<<<blocks((size_t)N * WID3, TPB), TPB, 0, stream>>>(
        bagg, cmean, crstd, bn3_g, bn3_b, nullptr, nullptr, bx, nullptr, N, WID3);

    // ============== Pooling ==============
    fill_i32<<<1, NGRP, 0, stream>>>(cnt, 0, NGRP);
    fill_f32<<<blocks((size_t)NGRP * WID3, TPB), TPB, 0, stream>>>(psum, 0.f, NGRP * WID3);
    fill_f32<<<blocks((size_t)NGRP * WID3, TPB), TPB, 0, stream>>>(pmax, -INFINITY, NGRP * WID3);
    pool_count<<<blocks(N, TPB), TPB, 0, stream>>>(batch, cnt, N);
    pool_acc<<<blocks((size_t)N * WID3, TPB), TPB, 0, stream>>>(bx, batch, psum, pmax, N, WID3);
    pool_build<<<blocks((size_t)NGRP * 2 * WID3, TPB), TPB, 0, stream>>>(psum, pmax, cnt, pooled, WID3);

    // ============== MLP head ==============
    mlp_gemm<<<blocks((size_t)NGRP * 256, TPB), TPB, 0, stream>>>(pooled, pW1, pb1, mh1, NGRP, 1024, 256, 1);
    mlp_bn<<<1, 256, 0, stream>>>(mh1, pbn1g, pbn1b, NGRP, 256);
    mlp_gemm<<<blocks((size_t)NGRP * 128, TPB), TPB, 0, stream>>>(mh1, pW2, pb2, mh2, NGRP, 256, 128, 1);
    mlp_bn<<<1, 128, 0, stream>>>(mh2, pbn2g, pbn2b, NGRP, 128);
    mlp_gemm<<<blocks((size_t)NGRP * 64, TPB), TPB, 0, stream>>>(mh2, pW3, pb3, mh3, NGRP, 128, 64, 1);
    mlp_gemm<<<1, TPB, 0, stream>>>(mh3, pW4, pb4, (float*)d_out, NGRP, 64, 1, 0);
}
